// GRU_31559419691392
// MI455X (gfx1250) — compile-verified, matrix operand order checked by reading, not run on hardware
//
#include <hip/hip_runtime.h>
#include <cstddef>

// ---------------------------------------------------------------------------
// Types for CDNA5 WMMA (wave32).
// ---------------------------------------------------------------------------
typedef __bf16 bf16_t;
typedef __attribute__((ext_vector_type(8)))  __bf16 bfx8;
typedef __attribute__((ext_vector_type(16))) __bf16 v16bf;
typedef __attribute__((ext_vector_type(8)))  float  v8f;

// Problem constants (match reference).
constexpr int B_ = 64, T_ = 2048, C_ = 256, H_ = 512, O_ = 256;
constexpr int NWG = 16;          // persistent workgroups (one WGP each)
constexpr float EPS_ = 1e-12f;

// ---------------------------------------------------------------------------
// Helpers
// ---------------------------------------------------------------------------
__device__ __forceinline__ v8f bwmma(v16bf a, v16bf b, v8f c) {
  return __builtin_amdgcn_wmma_f32_16x16x32_bf16(
      /*neg_a=*/false, a, /*neg_b=*/false, b,
      /*c_mod=*/(short)0, c, /*reuse_a=*/false, /*reuse_b=*/false);
}

// Load one 16-element bf16 fragment for A/B of v_wmma_f32_16x16x32_bf16.
// `p` must already point at  src + row*stride + kbase + (lane>>4)*8 .
// Per-lane layout (ISA 16-bit A-matrix table): halfs [0..7] at K+0..7,
// halfs [8..15] at K+16..23 (each lane-half offset by 8 via fh8).
__device__ __forceinline__ v16bf load_frag(const bf16_t* p) {
  bfx8 lo = *(const bfx8*)p;
  bfx8 hi = *(const bfx8*)(p + 16);
  v16bf f;
#pragma unroll
  for (int i = 0; i < 8; ++i) { f[i] = lo[i]; f[8 + i] = hi[i]; }
  return f;
}

__device__ __forceinline__ float sigm(float x) { return 1.f / (1.f + expf(-x)); }
__device__ __forceinline__ float softplus_(float x) {
  return fmaxf(x, 0.f) + log1pf(expf(-fabsf(x)));
}

// Device-wide sense barrier (monotonic target).  bar[0]=arrive, bar[1]=phase.
__device__ __forceinline__ void grid_sync(unsigned* bar, unsigned target) {
  __syncthreads();
  if (threadIdx.x == 0) {
    __threadfence();
    unsigned prev = __hip_atomic_fetch_add(&bar[0], 1u, __ATOMIC_ACQ_REL,
                                           __HIP_MEMORY_SCOPE_AGENT);
    if (prev == (unsigned)NWG - 1u) {
      __hip_atomic_store(&bar[0], 0u, __ATOMIC_RELAXED, __HIP_MEMORY_SCOPE_AGENT);
      __hip_atomic_store(&bar[1], target, __ATOMIC_RELEASE, __HIP_MEMORY_SCOPE_AGENT);
    } else {
      while (__hip_atomic_load(&bar[1], __ATOMIC_ACQUIRE,
                               __HIP_MEMORY_SCOPE_AGENT) < target)
        __builtin_amdgcn_s_sleep(2);
    }
  }
  __syncthreads();
}

// ---------------------------------------------------------------------------
// Kernel: zero the cross-kernel scratch (ws is poisoned by the harness).
// ---------------------------------------------------------------------------
__global__ void k_init(unsigned* bar, float* normacc, float* rownorm) {
  int i = threadIdx.x;
  if (i < 2)   bar[i] = 0u;
  if (i < 128) normacc[i] = 0.f;
  if (i < 64)  rownorm[i] = 0.f;
}

// ---------------------------------------------------------------------------
// Kernel: fp32 -> bf16 conversion of x_t (t=1..T-1) into (t-1,b,c) layout.
// ---------------------------------------------------------------------------
__global__ void k_xconv(const float* __restrict__ cv, unsigned short* Xbf_us) {
  bf16_t* Xbf = (bf16_t*)Xbf_us;
  const size_t total = (size_t)(T_ - 1) * B_ * C_;
  for (size_t i = (size_t)blockIdx.x * blockDim.x + threadIdx.x; i < total;
       i += (size_t)gridDim.x * blockDim.x) {
    size_t c   = i & (size_t)(C_ - 1);
    size_t row = i >> 8;            // (t-1)*B + b
    size_t tm1 = row >> 6;
    size_t bb  = row & 63u;
    Xbf[i] = (bf16_t)cv[(bb * T_ + (tm1 + 1)) * C_ + c];
  }
}

__global__ void k_wconv(const float* __restrict__ w_out, unsigned short* wo_us) {
  bf16_t* wo = (bf16_t*)wo_us;
  int i = blockIdx.x * blockDim.x + threadIdx.x;   // 256*512 threads exactly
  wo[i] = (bf16_t)w_out[i];
}

// ---------------------------------------------------------------------------
// MLP head for h0 (tiny: scalar fp32, negligible runtime).
// ---------------------------------------------------------------------------
__global__ void k_mlp0(const float* __restrict__ cv, const float* __restrict__ w0,
                       const float* __restrict__ b0, float* __restrict__ h1) {
  int o = blockIdx.x * blockDim.x + threadIdx.x;   // 64*512
  int bIdx = o >> 9, m = o & 511;
  const float* x = cv + (size_t)bIdx * T_ * C_;    // x0 = control[:,0,:]
  const float* w = w0 + (size_t)m * C_;
  float s = b0[m];
  for (int k = 0; k < C_; ++k) s += x[k] * w[k];
  h1[o] = softplus_(s);
}

__global__ void k_mlp1(const float* __restrict__ h1, const float* __restrict__ w1,
                       const float* __restrict__ b1, float* __restrict__ h2) {
  int o = blockIdx.x * blockDim.x + threadIdx.x;
  int bIdx = o >> 9, m = o & 511;
  const float* x = h1 + (size_t)bIdx * 512;
  const float* w = w1 + (size_t)m * 512;
  float s = b1[m];
  for (int k = 0; k < 512; ++k) s += x[k] * w[k];
  h2[o] = softplus_(s);
}

__global__ void k_mlp2(const float* __restrict__ h2, const float* __restrict__ w2,
                       const float* __restrict__ b2, float* __restrict__ h0f,
                       float* __restrict__ rownorm) {
  int o = blockIdx.x * blockDim.x + threadIdx.x;
  int bIdx = o >> 9, m = o & 511;
  const float* x = h2 + (size_t)bIdx * 512;
  const float* w = w2 + (size_t)m * 512;
  float s = b2[m];
  for (int k = 0; k < 512; ++k) s += x[k] * w[k];
  h0f[o] = s;
  atomicAdd(&rownorm[bIdx], s * s);
}

__global__ void k_mlp3(float* __restrict__ h0f, const float* __restrict__ rownorm,
                       unsigned short* hs_us) {
  bf16_t* hs = (bf16_t*)hs_us;                     // hs[0] block
  int o = blockIdx.x * blockDim.x + threadIdx.x;
  int bIdx = o >> 9;
  float v = h0f[o] * rsqrtf(rownorm[bIdx] + EPS_);
  h0f[o] = v;
  hs[o] = (bf16_t)v;
}

// ---------------------------------------------------------------------------
// Persistent recurrent kernel: 16 WGs x 256 threads, 2 grid barriers / step.
// WG `wg` owns h-columns [wg*32, wg*32+32).  LDS holds its bf16 weight slice:
//   r rows (merged K=768), z rows (K=768), n_x rows (K=256), n_h rows (K=512).
// ---------------------------------------------------------------------------
__global__ __launch_bounds__(256, 1) void k_recurrent(
    const float* __restrict__ w_ih, const float* __restrict__ w_hh,
    const float* __restrict__ bias, const float* __restrict__ b_n,
    const float* __restrict__ h0f, const unsigned short* __restrict__ Xbf_us,
    unsigned short* __restrict__ hs_us, float* normacc, unsigned* bar) {
  const bf16_t* Xbf = (const bf16_t*)Xbf_us;
  bf16_t* hs = (bf16_t*)hs_us;

  __shared__ bf16_t wlds[73728];        // 144 KB: 32*768(r)+32*768(z)+32*256+32*512
  __shared__ float  gbuf[64][128];      // 32 KB gate buffer: [r z nx nh] x 32 cols
  __shared__ float  hloc[64][32];       // fp32 h slice (persistent across steps)
  __shared__ float  msum[64];

  const int wg = blockIdx.x, tid = threadIdx.x;
  const int lane = tid & 31, wv = tid >> 5;
  const int fm = lane & 15;             // fragment row/col within tile
  const int fh8 = (lane >> 4) * 8;      // K sub-offset per lane half
  const int j0 = wg * 32;

  // ---- one-time: stage bf16 weight slice into LDS --------------------------
  for (int idx = tid; idx < 32 * 768; idx += 256) {
    int rr = idx / 768, k = idx - rr * 768;
    float vr = (k < 256) ? w_ih[(j0 + rr) * 256 + k]
                         : w_hh[(j0 + rr) * 512 + (k - 256)];
    float vz = (k < 256) ? w_ih[(512 + j0 + rr) * 256 + k]
                         : w_hh[(512 + j0 + rr) * 512 + (k - 256)];
    wlds[idx]          = (bf16_t)vr;
    wlds[24576 + idx]  = (bf16_t)vz;
  }
  for (int idx = tid; idx < 32 * 256; idx += 256) {
    int rr = idx >> 8, k = idx & 255;
    wlds[49152 + idx] = (bf16_t)w_ih[(1024 + j0 + rr) * 256 + k];
  }
  for (int idx = tid; idx < 32 * 512; idx += 256) {
    int rr = idx >> 9, k = idx & 511;
    wlds[57344 + idx] = (bf16_t)w_hh[(1024 + j0 + rr) * 512 + k];
  }
  for (int idx = tid; idx < 64 * 32; idx += 256)
    hloc[idx >> 5][idx & 31] = h0f[(idx >> 5) * 512 + j0 + (idx & 31)];
  __syncthreads();

  const int m  = tid >> 2;              // elementwise: 4 threads per batch row
  const int jb = (tid & 3) * 8;

  // ---- the scan ------------------------------------------------------------
  for (int t = 1; t < T_; ++t) {
    const bf16_t* xa = Xbf + (size_t)(t - 1) * (64 * 256);
    const bf16_t* ha = hs + (size_t)(t - 1) * (64 * 512);

    // 32 (mtile,ntile) jobs over 8 waves; ntile 0-1:r 2-3:z 4-5:n_x 6-7:n_h.
    for (int job = wv; job < 32; job += 8) {
      const int mt = job >> 3, nt = job & 7;
      v8f acc = {0.f, 0.f, 0.f, 0.f, 0.f, 0.f, 0.f, 0.f};
      const bf16_t* arow_x = xa + (mt * 16 + fm) * 256 + fh8;
      const bf16_t* arow_h = ha + (mt * 16 + fm) * 512 + fh8;
      if (nt < 4) {                                   // r / z : K = 256 + 512
        const int rbase = (nt < 2) ? nt * 16 : 32 + (nt - 2) * 16;
        const bf16_t* wb = &wlds[rbase * 768 + fm * 768 + fh8];
#pragma unroll 4
        for (int kk = 0; kk < 8; ++kk)
          acc = bwmma(load_frag(arow_x + kk * 32), load_frag(wb + kk * 32), acc);
#pragma unroll 4
        for (int kk = 0; kk < 16; ++kk)
          acc = bwmma(load_frag(arow_h + kk * 32),
                      load_frag(wb + 256 + kk * 32), acc);
      } else if (nt < 6) {                            // n_x : K = 256
        const bf16_t* wb = &wlds[49152 + (nt - 4) * 16 * 256 + fm * 256 + fh8];
#pragma unroll 4
        for (int kk = 0; kk < 8; ++kk)
          acc = bwmma(load_frag(arow_x + kk * 32), load_frag(wb + kk * 32), acc);
      } else {                                        // n_h : K = 512
        const bf16_t* wb = &wlds[57344 + (nt - 6) * 16 * 512 + fm * 512 + fh8];
#pragma unroll 4
        for (int kk = 0; kk < 16; ++kk)
          acc = bwmma(load_frag(arow_h + kk * 32), load_frag(wb + kk * 32), acc);
      }
      // C/D layout: VGPR r -> M = mt*16 + fh8 + r, N = nt*16 + fm.
#pragma unroll
      for (int r = 0; r < 8; ++r)
        gbuf[mt * 16 + fh8 + r][nt * 16 + fm] = acc[r];
    }

    if (t < T_ - 1)   // warm L2/L0 for next step's x tile
      __builtin_prefetch(Xbf + (size_t)t * (64 * 256) + tid * 64, 0, 1);

    __syncthreads();
    if (tid < 64) msum[tid] = 0.f;
    __syncthreads();

    // ---- elementwise gates + partial norm ----------------------------------
    {
      float s = 0.f;
#pragma unroll
      for (int q = 0; q < 8; ++q) {
        int jj = jb + q, j = j0 + jj;
        float r  = sigm(gbuf[m][jj]       + bias[j]);
        float z  = sigm(gbuf[m][32 + jj]  + bias[512 + j]);
        float n  = tanhf(gbuf[m][64 + jj] + bias[1024 + j] +
                         r * (gbuf[m][96 + jj] + b_n[j]));
        float h   = hloc[m][jj];
        float pre = n + z * (h - n);
        gbuf[m][64 + jj] = pre;          // reuse nx slot as pre-buffer
        s += pre * pre;
      }
      atomicAdd(&msum[m], s);
    }
    __syncthreads();
    if (tid < 64) atomicAdd(&normacc[(t & 1) * 64 + tid], msum[tid]);

    grid_sync(bar, 2u * (unsigned)t - 1u);           // norm complete

    // ---- retract + publish h_t ---------------------------------------------
    {
      float nrm = __hip_atomic_load(&normacc[(t & 1) * 64 + m],
                                    __ATOMIC_RELAXED, __HIP_MEMORY_SCOPE_AGENT);
      float sc = rsqrtf(nrm + EPS_);
      bf16_t* hw = hs + (size_t)t * (64 * 512) + m * 512 + j0 + jb;
#pragma unroll
      for (int q = 0; q < 8; ++q) {
        float hn = gbuf[m][64 + jb + q] * sc;
        hloc[m][jb + q] = hn;
        hw[q] = (bf16_t)hn;
      }
    }
    if (wg == 0 && tid < 64) normacc[((t + 1) & 1) * 64 + tid] = 0.f;

    grid_sync(bar, 2u * (unsigned)t);                // h_t visible
  }
}

// ---------------------------------------------------------------------------
// Output projection: y = retract(hs @ w_out^T + b_out), parallel over T*B rows.
// One wave per 16-row m-tile; 16 n-tiles of 16 cols; K=512 (16 bf16 chunks).
// ---------------------------------------------------------------------------
__global__ __launch_bounds__(256) void k_out(
    const unsigned short* __restrict__ hs_us,
    const unsigned short* __restrict__ wo_us,
    const float* __restrict__ b_out, float* __restrict__ out) {
  const bf16_t* hs = (const bf16_t*)hs_us;
  const bf16_t* wo = (const bf16_t*)wo_us;
  __shared__ float obuf[8][16][256];    // 128 KB
  __shared__ float snorm[8][16];

  const int tid = threadIdx.x, lane = tid & 31, wv = tid >> 5;
  const int fm = lane & 15, fh8 = (lane >> 4) * 8;
  const int row0 = (blockIdx.x * 8 + wv) * 16;       // row in (t,b) order
  const bf16_t* arow = hs + (size_t)(row0 + fm) * 512 + fh8;

  for (int nt = 0; nt < 16; ++nt) {
    v8f acc = {0.f, 0.f, 0.f, 0.f, 0.f, 0.f, 0.f, 0.f};
    const bf16_t* wb = wo + (nt * 16 + fm) * 512 + fh8;
#pragma unroll 4
    for (int kk = 0; kk < 16; ++kk)
      acc = bwmma(load_frag(arow + kk * 32), load_frag(wb + kk * 32), acc);
    float bo = b_out[nt * 16 + fm];
#pragma unroll
    for (int r = 0; r < 8; ++r)
      obuf[wv][fh8 + r][nt * 16 + fm] = acc[r] + bo;
  }
  __syncthreads();

  if (lane < 16) {
    float s = 0.f;
    for (int o = 0; o < 256; ++o) { float v = obuf[wv][lane][o]; s += v * v; }
    snorm[wv][lane] = rsqrtf(s + EPS_);
  }
  __syncthreads();

  for (int e = lane; e < 16 * 256; e += 32) {
    int r = e >> 8, o = e & 255;
    int grow = row0 + r;                 // = t*64 + b
    int t = grow >> 6, bb = grow & 63;
    out[((size_t)bb * T_ + t) * O_ + o] = obuf[wv][r][o] * snorm[wv][r];
  }
}

// ---------------------------------------------------------------------------
// Launch
// ---------------------------------------------------------------------------
extern "C" void kernel_launch(void* const* d_in, const int* in_sizes, int n_in,
                              void* d_out, int out_size, void* d_ws, size_t ws_size,
                              hipStream_t stream) {
  (void)in_sizes; (void)n_in; (void)out_size;
  const float* cv    = (const float*)d_in[0];
  const float* w0    = (const float*)d_in[1];
  const float* b0    = (const float*)d_in[2];
  const float* w1    = (const float*)d_in[3];
  const float* b1    = (const float*)d_in[4];
  const float* w2    = (const float*)d_in[5];
  const float* b2    = (const float*)d_in[6];
  const float* w_ih  = (const float*)d_in[7];
  const float* w_hh  = (const float*)d_in[8];
  const float* bias  = (const float*)d_in[9];
  const float* b_n   = (const float*)d_in[10];
  const float* w_out = (const float*)d_in[11];
  const float* b_out = (const float*)d_in[12];
  float* out = (float*)d_out;

  // Workspace layout (all 256B-aligned offsets).
  char* ws = (char*)d_ws;
  size_t off = 0;
  const size_t HS_BYTES  = (size_t)T_ * B_ * H_ * 2;          // 134,217,728
  const size_t XBF_BYTES = (size_t)(T_ - 1) * B_ * C_ * 2;    //  67,076,096
  unsigned short* hs  = (unsigned short*)(ws + off); off += HS_BYTES;
  unsigned short* Xbf = (unsigned short*)(ws + off); off += XBF_BYTES;
  unsigned short* Wob = (unsigned short*)(ws + off); off += (size_t)O_ * H_ * 2;
  float* h0f     = (float*)(ws + off); off += (size_t)B_ * H_ * 4;
  float* h1      = (float*)(ws + off); off += (size_t)B_ * 512 * 4;
  float* h2      = (float*)(ws + off); off += (size_t)B_ * 512 * 4;
  float* normacc = (float*)(ws + off); off += 512;
  float* rownorm = (float*)(ws + off); off += 256;
  unsigned* bar  = (unsigned*)(ws + off); off += 256;
  if (off > ws_size) return;   // insufficient scratch; nothing sane to do

  k_init<<<1, 256, 0, stream>>>(bar, normacc, rownorm);
  k_xconv<<<4096, 256, 0, stream>>>(cv, Xbf);
  k_wconv<<<512, 256, 0, stream>>>(w_out, Wob);
  k_mlp0<<<128, 256, 0, stream>>>(cv, w0, b0, h1);
  k_mlp1<<<128, 256, 0, stream>>>(h1, w1, b1, h2);
  k_mlp2<<<128, 256, 0, stream>>>(h2, w2, b2, h0f, rownorm);
  k_mlp3<<<128, 256, 0, stream>>>(h0f, rownorm, hs);
  k_recurrent<<<NWG, 256, 0, stream>>>(w_ih, w_hh, bias, b_n, h0f, Xbf, hs,
                                       normacc, bar);
  k_out<<<1024, 256, 0, stream>>>(hs, Wob, b_out, out);
}
